// MODEL_GNN_15908558864830
// MI455X (gfx1250) — matrix operand order repaired and checked by reading, not compiled, not run on hardware
//
#include <hip/hip_runtime.h>
#include <hip/hip_bf16.h>
#include <hip/hip_fp16.h>

typedef __attribute__((ext_vector_type(16))) _Float16 v16h;
typedef __attribute__((ext_vector_type(8)))  float    v8f;

#define DEVINL __device__ __forceinline__

static constexpr int       NN = 262144;    // nodes
static constexpr long long EE = 4194304;   // edges
static constexpr int       GG = 512;       // graphs

// ---------------- workspace layout (bytes) ----------------
static constexpr size_t LAYER_PACK_BYTES = 5632;            // 5*512 f16 B-frags + 5*16 f32 bias + 17 f32 mlp7
static constexpr size_t OFF_WPACK = 0;                       // 4 * 5632
static constexpr size_t OFF_HA    = 32768;                   // N*16 f32
static constexpr size_t OFF_HB    = OFF_HA  + (size_t)NN*16*4;
static constexpr size_t OFF_AGG   = OFF_HB  + (size_t)NN*16*4;
static constexpr size_t OFF_NMAX  = OFF_AGG + (size_t)NN*16*4;   // N u32
static constexpr size_t OFF_NSUM  = OFF_NMAX + (size_t)NN*4;     // N f32
static constexpr size_t OFF_WE    = OFF_NSUM + (size_t)NN*4;     // E f32
static constexpr size_t OFF_MSG   = OFF_WE   + (size_t)EE*4;     // E*16 f16
static constexpr size_t OFF_GARR  = OFF_MSG  + (size_t)EE*16*2;  // N f32
static constexpr size_t OFF_GMAX  = OFF_GARR + (size_t)NN*4;     // G u32 (padded 4KB)
static constexpr size_t OFF_GSUM  = OFF_GMAX + 4096;             // G f32 (padded 4KB)
static constexpr size_t OFF_X1    = OFF_GSUM + 4096;             // G*8 f32
static constexpr size_t OFF_X2    = OFF_X1   + (size_t)GG*8*4;   // G*16 f32

static constexpr unsigned ENC_NEG_INF = 0x007FFFFFu; // enc(-inf)

// ---------------- device helpers ----------------
DEVINL unsigned enc_f32(float f) {
    unsigned u = __float_as_uint(f);
    return (u & 0x80000000u) ? ~u : (u | 0x80000000u);
}
DEVINL float dec_f32(unsigned k) {
    unsigned u = (k & 0x80000000u) ? (k ^ 0x80000000u) : ~k;
    return __uint_as_float(u);
}
DEVINL void wave_lds_fence() {
    __builtin_amdgcn_wave_barrier();
    asm volatile("s_wait_dscnt 0" ::: "memory");
    __builtin_amdgcn_wave_barrier();
}
DEVINL v8f zero8() {
    v8f z;
#pragma unroll
    for (int i = 0; i < 8; ++i) z[i] = 0.f;
    return z;
}
DEVINL v8f wmma_f16(v16h a, v16h b, v8f c) {
    return __builtin_amdgcn_wmma_f32_16x16x32_f16(false, a, false, b, (short)0, c, false, false);
}

union Frag16 { uint4 u[2]; v16h v; };

// A fragment from LDS row-major [16][32] halves: per lane it is exactly two
// contiguous 16B chunks (k = koff..koff+7 and 16+koff..16+koff+7).
DEVINL v16h load_Afrag(const _Float16* A, int lane) {
    const int row  = lane & 15;
    const int koff = (lane & 16) ? 8 : 0;
    const uint4* p = (const uint4*)(A + row * 32 + koff);
    Frag16 f;
    f.u[0] = p[0];   // k = koff .. koff+7
    f.u[1] = p[2];   // k = 16+koff .. 16+koff+23? no: +16 halves => k = 16+koff..16+koff+7
    return f.v;
}
// B fragment pre-swizzled in global memory: 32 lanes x 16 halves, contiguous per lane.
DEVINL v16h load_Bfrag(const _Float16* W, int lane) {
    const uint4* p = (const uint4*)(W + lane * 16);
    Frag16 f;
    f.u[0] = p[0];
    f.u[1] = p[1];
    return f.v;
}

// ---------------- parameter plumbing ----------------
// per-conv leaves (jax sorted-key order): e1w,e1b,e2w,e2b,m1w,m1b,m2w,m2b,m3w,m3b,m4w,m4b,m5w,m5b,m6w,m6b,m7w,m7b
struct ConvPtrs { const float* p[18]; };
struct AllConv  { ConvPtrs c[4]; };

// Convert conv weights into pre-swizzled per-lane B fragments + padded biases.
// frag[lane*16+h] = W[k*fout + col], k = (lane&16 ? 8:0)+(h<8 ? h : h+8), col = lane&15, zero-padded.
__global__ void __launch_bounds__(256) prep_weights(AllConv ap, char* wsbase) {
    const int l = blockIdx.x / 5, m = blockIdx.x % 5;
    const int cins[4]  = {1, 4, 8, 8};
    const int couts[4] = {4, 8, 8, 16};
    const ConvPtrs cp = ap.c[l];
    int fin, fout; const float* w; const float* b;
    switch (m) {
        case 0:  fin = 3;                fout = 16;       w = cp.p[0];  b = cp.p[1];  break;  // e1
        case 1:  fin = 16;               fout = 16;       w = cp.p[2];  b = cp.p[3];  break;  // e2
        case 2:  fin = 2 * cins[l] + 16; fout = couts[l]; w = cp.p[4];  b = cp.p[5];  break;  // mlp1
        case 3:  fin = cins[l];          fout = 16;       w = cp.p[12]; b = cp.p[13]; break;  // mlp5
        default: fin = couts[l];         fout = 16;       w = cp.p[14]; b = cp.p[15]; break;  // mlp6
    }
    char* lp = wsbase + OFF_WPACK + (size_t)l * LAYER_PACK_BYTES;
    _Float16* mats   = (_Float16*)lp;
    float*    biases = (float*)(lp + 5 * 512 * 2);
    float*    m7     = biases + 5 * 16;
    _Float16* om = mats + m * 512;
    for (int i = threadIdx.x; i < 512; i += 256) {
        const int lane = i >> 4, h = i & 15;
        const int c = lane & 15;
        const int k = ((lane & 16) ? 8 : 0) + (h < 8 ? h : h + 8);
        om[i] = (k < fin && c < fout) ? (_Float16)w[k * fout + c] : (_Float16)0.f;
    }
    if (threadIdx.x < 16) biases[m * 16 + threadIdx.x] = (threadIdx.x < fout) ? b[threadIdx.x] : 0.f;
    if (m == 0 && threadIdx.x < 17)
        m7[threadIdx.x] = (threadIdx.x < 16) ? cp.p[16][threadIdx.x] : cp.p[17][0];
}

__global__ void fill_u32(unsigned* p, unsigned v, long long n) {
    long long i = (long long)blockIdx.x * blockDim.x + threadIdx.x;
    if (i < n) p[i] = v;
}

// ---------------- edge pass A: all per-edge GEMMs via WMMA ----------------
template <int CIN, int COUT>
__global__ void __launch_bounds__(256) edge_pass_a(
    const int* __restrict__ srcp, const int* __restrict__ dstp,
    const float* __restrict__ eattr, const float* __restrict__ h_in,
    const _Float16* __restrict__ mats, const float* __restrict__ biases,
    const float* __restrict__ m7,
    _Float16* __restrict__ msg_buf, float* __restrict__ w_e,
    unsigned* __restrict__ node_max_u)
{
    __shared__ __attribute__((aligned(16))) _Float16 sA[8][16 * 32];
    __shared__ __attribute__((aligned(16))) float    sP[8][16 * 16];

    // one-time zero of the A tiles so first-iteration garbage is finite
    {
        uint4 z; z.x = 0u; z.y = 0u; z.z = 0u; z.w = 0u;
        uint4* pz = (uint4*)&sA[0][0];
        for (int i = threadIdx.x; i < (8 * 16 * 32 * 2) / 16; i += 256) pz[i] = z;
    }
    __syncthreads();

    const int wv   = threadIdx.x >> 5;
    const int lane = threadIdx.x & 31;
    const int col  = lane & 15;
    const int roff = (lane & 16) ? 8 : 0;
    _Float16* A = sA[wv];
    float*    P = sP[wv];

    // loop-invariant operands: B fragments straight from global (2x b128 each),
    // per-lane biases, uniform mlp7 weights (scalar loads)
    const v16h Bf_e1 = load_Bfrag(mats + 0 * 512, lane);
    const v16h Bf_e2 = load_Bfrag(mats + 1 * 512, lane);
    const v16h Bf_m1 = load_Bfrag(mats + 2 * 512, lane);
    const v16h Bf_m5 = load_Bfrag(mats + 3 * 512, lane);
    const v16h Bf_m6 = load_Bfrag(mats + 4 * 512, lane);
    const float b_e1 = biases[0 * 16 + col];
    const float b_e2 = biases[1 * 16 + col];
    const float b_m1 = biases[2 * 16 + col];
    const float b_m5 = biases[3 * 16 + col];
    const float b_m6 = biases[4 * 16 + col];
    float m7r[17];
#pragma unroll
    for (int i = 0; i < 17; ++i) m7r[i] = m7[i];

    const long long ntiles = EE / 16;
    const long long nwaves = (long long)gridDim.x * 8;
    for (long long tile = (long long)blockIdx.x * 8 + wv; tile < ntiles; tile += nwaves) {
        const long long e0 = tile * 16;
        int se = 0, de = 0;
        if (lane < 16) { se = srcp[e0 + lane]; de = dstp[e0 + lane]; }

        // ---- edge MLP stage 1: relu(attr @ e1 + b1)  (e1 zero-padded for k>=3) ----
        if (lane < 16) {
            const float* ap = eattr + (e0 + lane) * 3;
            A[lane * 32 + 0] = (_Float16)ap[0];
            A[lane * 32 + 1] = (_Float16)ap[1];
            A[lane * 32 + 2] = (_Float16)ap[2];
        }
        wave_lds_fence();
        v16h a = load_Afrag(A, lane);
        v8f  c = wmma_f16(a, Bf_e1, zero8());
        float t[8];
#pragma unroll
        for (int r = 0; r < 8; ++r) t[r] = fmaxf(c[r] + b_e1, 0.f);

        // ---- edge MLP stage 2: t @ e2 + b2  (e2 zero-padded for k>=16) ----
        wave_lds_fence();
#pragma unroll
        for (int r = 0; r < 8; ++r) A[(r + roff) * 32 + col] = (_Float16)t[r];
        wave_lds_fence();
        a = load_Afrag(A, lane);
        c = wmma_f16(a, Bf_e2, zero8());
        float ea[8];
#pragma unroll
        for (int r = 0; r < 8; ++r) ea[r] = c[r] + b_e2;

        // ---- build edge_input = [x_i, x_j - x_i, ea]; mlp1 zero-padded for k>=2*CIN+16 ----
        wave_lds_fence();
#pragma unroll
        for (int r = 0; r < 8; ++r) A[(r + roff) * 32 + 2 * CIN + col] = (_Float16)ea[r];
        if (lane < 16) {
            const float* hd = h_in + (long long)de * CIN;
            const float* hs = h_in + (long long)se * CIN;
            union H8 { _Float16 h[8]; uint2 u2; uint4 u4; } xi, dx;
#pragma unroll
            for (int f = 0; f < CIN; ++f) {
                const float vi = hd[f], vj = hs[f];
                xi.h[f] = (_Float16)vi;
                dx.h[f] = (_Float16)(vj - vi);
            }
            if constexpr (CIN == 8) {
                *(uint4*)(A + lane * 32)     = xi.u4;
                *(uint4*)(A + lane * 32 + 8) = dx.u4;
            } else if constexpr (CIN == 4) {
                *(uint2*)(A + lane * 32)     = xi.u2;
                *(uint2*)(A + lane * 32 + 4) = dx.u2;
            } else {
                A[lane * 32 + 0] = xi.h[0];
                A[lane * 32 + 1] = dx.h[0];
            }
        }
        wave_lds_fence();

        // ---- mlp1: msg = relu(edge_input @ m1 + b) ----
        a = load_Afrag(A, lane);
        c = wmma_f16(a, Bf_m1, zero8());
        float msg[8];
#pragma unroll
        for (int r = 0; r < 8; ++r) msg[r] = fmaxf(c[r] + b_m1, 0.f);

        // ---- mlp5: w1 = tanh(x_i @ m5 + b5); masked fragment derived in registers ----
        // k < CIN (<=8) holds x_i and maps to low lanes, h < CIN only.
        v16h a5;
        {
            const bool lo = (lane < 16);
#pragma unroll
            for (int h = 0; h < 16; ++h)
                a5[h] = (lo && h < CIN) ? a[h] : (_Float16)0.f;
        }
        c = wmma_f16(a5, Bf_m5, zero8());
        float w1[8];
#pragma unroll
        for (int r = 0; r < 8; ++r) w1[r] = tanhf(c[r] + b_m5);

        // ---- stage msg tile (A layout); mlp6 zero-padded for k>=COUT ----
        wave_lds_fence();
#pragma unroll
        for (int r = 0; r < 8; ++r)
            A[(r + roff) * 32 + col] = (col < COUT) ? (_Float16)msg[r] : (_Float16)0.f;
        wave_lds_fence();
        // vectorized msg row store to global f16
        if (lane < 16) {
            _Float16*       dm = msg_buf + (e0 + lane) * COUT;
            const _Float16* sr = A + lane * 32;
            if constexpr (COUT == 16) {
                ((uint4*)dm)[0] = ((const uint4*)sr)[0];
                ((uint4*)dm)[1] = ((const uint4*)sr)[1];
            } else if constexpr (COUT == 8) {
                ((uint4*)dm)[0] = ((const uint4*)sr)[0];
            } else {
                ((uint2*)dm)[0] = ((const uint2*)sr)[0];
            }
        }

        // ---- mlp6: w2 = tanh(msg @ m6 + b6); P = w1*w2 ----
        a = load_Afrag(A, lane);
        c = wmma_f16(a, Bf_m6, zero8());
#pragma unroll
        for (int r = 0; r < 8; ++r)
            P[(r + roff) * 16 + col] = w1[r] * tanhf(c[r] + b_m6);
        wave_lds_fence();

        // ---- w = (w1*w2) @ m7 + b7 ; softmax-max atomic keyed by src ----
        if (lane < 16) {
            const float4* pr = (const float4*)(P + lane * 16);
            const float4 q0 = pr[0], q1 = pr[1], q2 = pr[2], q3 = pr[3];
            float ww = m7r[16];
            ww += q0.x * m7r[0]  + q0.y * m7r[1]  + q0.z * m7r[2]  + q0.w * m7r[3];
            ww += q1.x * m7r[4]  + q1.y * m7r[5]  + q1.z * m7r[6]  + q1.w * m7r[7];
            ww += q2.x * m7r[8]  + q2.y * m7r[9]  + q2.z * m7r[10] + q2.w * m7r[11];
            ww += q3.x * m7r[12] + q3.y * m7r[13] + q3.z * m7r[14] + q3.w * m7r[15];
            w_e[e0 + lane] = ww;
            atomicMax(node_max_u + se, enc_f32(ww));
        }
        wave_lds_fence();
    }
}

// ---------------- edge pass B: exp + segment sum (keyed by src) ----------------
__global__ void edge_pass_b(const int* __restrict__ srcp, float* __restrict__ w_e,
                            const unsigned* __restrict__ nmax, float* __restrict__ nsum) {
    long long e = (long long)blockIdx.x * 256 + threadIdx.x;
    if (e >= EE) return;
    const int   s  = srcp[e];
    const float we = __expf(w_e[e] - dec_f32(nmax[s]));
    w_e[e] = we;
    atomicAdd(nsum + s, we);
}

// ---------------- edge pass C: alpha-weighted scatter to dst ----------------
template <int COUT>
__global__ void edge_pass_c(const int* __restrict__ srcp, const int* __restrict__ dstp,
                            const float* __restrict__ w_e, const float* __restrict__ nsum,
                            const _Float16* __restrict__ msg_buf, float* __restrict__ agg) {
    long long e = (long long)blockIdx.x * 256 + threadIdx.x;
    if (e >= EE) return;
    const int   s = srcp[e], d = dstp[e];
    const float alpha = w_e[e] / (nsum[s] + 1e-16f);
    const _Float16* mr = msg_buf + e * COUT;
#pragma unroll
    for (int f = 0; f < COUT; ++f)
        atomicAdd(agg + (long long)d * COUT + f, alpha * (float)mr[f]);
}

// ---------------- node update: gates + residual mix + relu ----------------
template <int CIN, int COUT>
__global__ void node_update(const float* __restrict__ h_in, const float* __restrict__ agg,
                            const float* __restrict__ m2w, const float* __restrict__ m2b,
                            const float* __restrict__ m3w, const float* __restrict__ m3b,
                            const float* __restrict__ m4w, const float* __restrict__ m4b,
                            float* __restrict__ h_out) {
    const int n = blockIdx.x * 256 + threadIdx.x;
    if (n >= NN) return;
    float xin[CIN];
#pragma unroll
    for (int f = 0; f < CIN; ++f) xin[f] = h_in[(long long)n * CIN + f];
    float xt[COUT], ag[COUT];
#pragma unroll
    for (int co = 0; co < COUT; ++co) {
        float s = m2b[co];
#pragma unroll
        for (int f = 0; f < CIN; ++f) s += xin[f] * m2w[f * COUT + co];
        xt[co] = fmaxf(s, 0.f);
        ag[co] = agg[(long long)n * COUT + co];
    }
    float z1 = m3b[0], z2 = m4b[0];
#pragma unroll
    for (int co = 0; co < COUT; ++co) {
        z1 += xt[co] * m3w[co] + ag[co] * m3w[COUT + co];
        z2 += xt[co] * m4w[co] + ag[co] * m4w[COUT + co];
    }
    const float g1 = 1.f / (1.f + __expf(-z1));
    const float g2 = 1.f / (1.f + __expf(-z2));
#pragma unroll
    for (int co = 0; co < COUT; ++co)
        h_out[(long long)n * COUT + co] = fmaxf(g1 * ag[co] + g2 * xt[co], 0.f);
}

// ---------------- gated global attention pooling ----------------
template <int C>
__global__ void gap_gate(const float* __restrict__ h, const int* __restrict__ batch,
                         const float* __restrict__ gw, const float* __restrict__ gb,
                         float* __restrict__ garr, unsigned* __restrict__ gmax) {
    const int n = blockIdx.x * 256 + threadIdx.x;
    if (n >= NN) return;
    float g = gb[0];
#pragma unroll
    for (int c = 0; c < C; ++c) g += h[(long long)n * C + c] * gw[c];
    garr[n] = g;
    atomicMax(gmax + batch[n], enc_f32(g));
}
__global__ void gap_exp(const int* __restrict__ batch, float* __restrict__ garr,
                        const unsigned* __restrict__ gmax, float* __restrict__ gsum) {
    const int n = blockIdx.x * 256 + threadIdx.x;
    if (n >= NN) return;
    const int   bg = batch[n];
    const float ge = __expf(garr[n] - dec_f32(gmax[bg]));
    garr[n] = ge;
    atomicAdd(gsum + bg, ge);
}
template <int C>
__global__ void gap_scat(const float* __restrict__ h, const int* __restrict__ batch,
                         const float* __restrict__ garr, const float* __restrict__ gsum,
                         float* __restrict__ xout) {
    const int n = blockIdx.x * 256 + threadIdx.x;
    if (n >= NN) return;
    const int   bg    = batch[n];
    const float alpha = garr[n] / (gsum[bg] + 1e-16f);
#pragma unroll
    for (int c = 0; c < C; ++c)
        atomicAdd(xout + (long long)bg * C + c, alpha * h[(long long)n * C + c]);
}

// ---------------- final MLP: [24]->32->16->16->1 ----------------
__global__ void final_mlp(const float* __restrict__ x1, const float* __restrict__ x2,
                          const float* __restrict__ w1, const float* __restrict__ b1,
                          const float* __restrict__ w2, const float* __restrict__ b2,
                          const float* __restrict__ w3, const float* __restrict__ b3,
                          const float* __restrict__ w4, const float* __restrict__ b4,
                          float* __restrict__ out) {
    const int g = blockIdx.x * 256 + threadIdx.x;
    if (g >= GG) return;
    float z[24];
#pragma unroll
    for (int i = 0; i < 8; ++i)  z[i]     = x1[g * 8 + i];
#pragma unroll
    for (int i = 0; i < 16; ++i) z[8 + i] = x2[g * 16 + i];
    float a1[32];
#pragma unroll
    for (int o = 0; o < 32; ++o) {
        float s = b1[o];
#pragma unroll
        for (int i = 0; i < 24; ++i) s += z[i] * w1[i * 32 + o];
        a1[o] = fmaxf(s, 0.f);
    }
    float a2[16];
#pragma unroll
    for (int o = 0; o < 16; ++o) {
        float s = b2[o];
#pragma unroll
        for (int i = 0; i < 32; ++i) s += a1[i] * w2[i * 16 + o];
        a2[o] = fmaxf(s, 0.f);
    }
    float a3[16];
#pragma unroll
    for (int o = 0; o < 16; ++o) {
        float s = b3[o];
#pragma unroll
        for (int i = 0; i < 16; ++i) s += a2[i] * w3[i * 16 + o];
        a3[o] = fmaxf(s, 0.f);
    }
    float s = b4[0];
#pragma unroll
    for (int i = 0; i < 16; ++i) s += a3[i] * w4[i];
    out[g] = s;
}

// ---------------- host orchestration ----------------
static inline int blocks_for(long long n) { return (int)((n + 255) / 256); }

template <int CIN, int COUT>
static void run_conv(const ConvPtrs& cp, int layer, const float* h_in, float* h_out,
                     const int* srcp, const int* dstp, const float* eattr,
                     char* ws, hipStream_t stream) {
    float*    agg  = (float*)(ws + OFF_AGG);
    unsigned* nmax = (unsigned*)(ws + OFF_NMAX);
    float*    nsum = (float*)(ws + OFF_NSUM);
    float*    w_e  = (float*)(ws + OFF_WE);
    _Float16* msg  = (_Float16*)(ws + OFF_MSG);
    char* lp = ws + OFF_WPACK + (size_t)layer * LAYER_PACK_BYTES;
    const _Float16* mats   = (const _Float16*)lp;
    const float*    biases = (const float*)(lp + 5 * 512 * 2);
    const float*    m7     = biases + 5 * 16;

    fill_u32<<<blocks_for(NN), 256, 0, stream>>>(nmax, ENC_NEG_INF, NN);
    fill_u32<<<blocks_for(NN), 256, 0, stream>>>((unsigned*)nsum, 0u, NN);
    fill_u32<<<blocks_for((long long)NN * COUT), 256, 0, stream>>>((unsigned*)agg, 0u, (long long)NN * COUT);

    edge_pass_a<CIN, COUT><<<2048, 256, 0, stream>>>(srcp, dstp, eattr, h_in, mats, biases, m7,
                                                     msg, w_e, nmax);
    edge_pass_b<<<(int)(EE / 256), 256, 0, stream>>>(srcp, w_e, nmax, nsum);
    edge_pass_c<COUT><<<(int)(EE / 256), 256, 0, stream>>>(srcp, dstp, w_e, nsum, msg, agg);
    node_update<CIN, COUT><<<blocks_for(NN), 256, 0, stream>>>(
        h_in, agg, cp.p[6], cp.p[7], cp.p[8], cp.p[9], cp.p[10], cp.p[11], h_out);
}

template <int C>
static void run_gap(const float* h, const int* batch, const float* gw, const float* gb,
                    float* xout, char* ws, hipStream_t stream) {
    float*    garr = (float*)(ws + OFF_GARR);
    unsigned* gmax = (unsigned*)(ws + OFF_GMAX);
    float*    gsum = (float*)(ws + OFF_GSUM);
    fill_u32<<<blocks_for(GG), 256, 0, stream>>>(gmax, ENC_NEG_INF, GG);
    fill_u32<<<blocks_for(GG), 256, 0, stream>>>((unsigned*)gsum, 0u, GG);
    fill_u32<<<blocks_for((long long)GG * C), 256, 0, stream>>>((unsigned*)xout, 0u, (long long)GG * C);
    gap_gate<C><<<blocks_for(NN), 256, 0, stream>>>(h, batch, gw, gb, garr, gmax);
    gap_exp<<<blocks_for(NN), 256, 0, stream>>>(batch, garr, gmax, gsum);
    gap_scat<C><<<blocks_for(NN), 256, 0, stream>>>(h, batch, garr, gsum, xout);
}

extern "C" void kernel_launch(void* const* d_in, const int* in_sizes, int n_in,
                              void* d_out, int out_size, void* d_ws, size_t ws_size,
                              hipStream_t stream) {
    (void)in_sizes; (void)n_in; (void)out_size; (void)ws_size;

    const float* x     = (const float*)d_in[0];
    const int*   ei    = (const int*)d_in[1];
    const int*   srcp  = ei;
    const int*   dstp  = ei + EE;
    const float* eattr = (const float*)d_in[2];
    const int*   batch = (const int*)d_in[3];

    // params flattened jax-tree style (sorted dict keys): conv1..4, gate1, gate2, lin1..4
    AllConv ap;
    for (int l = 0; l < 4; ++l)
        for (int j = 0; j < 18; ++j)
            ap.c[l].p[j] = (const float*)d_in[4 + 18 * l + j];
    const float* g1w = (const float*)d_in[76];
    const float* g1b = (const float*)d_in[77];
    const float* g2w = (const float*)d_in[78];
    const float* g2b = (const float*)d_in[79];
    const float* l1w = (const float*)d_in[80];
    const float* l1b = (const float*)d_in[81];
    const float* l2w = (const float*)d_in[82];
    const float* l2b = (const float*)d_in[83];
    const float* l3w = (const float*)d_in[84];
    const float* l3b = (const float*)d_in[85];
    const float* l4w = (const float*)d_in[86];
    const float* l4b = (const float*)d_in[87];

    char*  ws = (char*)d_ws;
    float* hA = (float*)(ws + OFF_HA);
    float* hB = (float*)(ws + OFF_HB);
    float* x1 = (float*)(ws + OFF_X1);
    float* x2 = (float*)(ws + OFF_X2);

    prep_weights<<<20, 256, 0, stream>>>(ap, ws);

    run_conv<1, 4>(ap.c[0], 0, x,  hA, srcp, dstp, eattr, ws, stream);
    run_conv<4, 8>(ap.c[1], 1, hA, hB, srcp, dstp, eattr, ws, stream);
    run_gap<8>(hB, batch, g1w, g1b, x1, ws, stream);
    run_conv<8, 8>(ap.c[2], 2, hB, hA, srcp, dstp, eattr, ws, stream);
    run_conv<8, 16>(ap.c[3], 3, hA, hB, srcp, dstp, eattr, ws, stream);
    run_gap<16>(hB, batch, g2w, g2b, x2, ws, stream);

    final_mlp<<<2, 256, 0, stream>>>(x1, x2, l1w, l1b, l2w, l2b, l3w, l3b, l4w, l4b,
                                     (float*)d_out);
}
